// SWA_77395310674145
// MI455X (gfx1250) — matrix operand order repaired and checked by reading, compiled, not run on hardware
//
#include <hip/hip_runtime.h>
#include <hip/hip_bf16.h>
#include <stdint.h>

#define B_     2
#define Q_     512
#define HQ_    32
#define HKV_   8
#define D_     128
#define BS_    128
#define NBLK_  16
#define MAXK_  (BS_ * NBLK_)
#define GQA_   (HQ_ / HKV_)   /* 4 */
#define QCHUNK_ 32
#define NEG_   (-1e30f)

typedef __attribute__((ext_vector_type(16))) __bf16 v16bf;
typedef __attribute__((ext_vector_type(8)))  float  v8f;

union ABu { v16bf v; unsigned int u[8]; };
union Cu  { v8f   v; float        f[8]; };

__device__ __forceinline__ unsigned int pack_bf16(float lof, float hif) {
    unsigned int a = __builtin_bit_cast(unsigned int, lof);
    unsigned int b = __builtin_bit_cast(unsigned int, hif);
    a += 0x7FFFu + ((a >> 16) & 1u);   // round-to-nearest-even
    b += 0x7FFFu + ((b >> 16) & 1u);
    return (a >> 16) | (b & 0xFFFF0000u);
}

__global__ __launch_bounds__(256) void swa_fwd_kernel(
    const float* __restrict__ qp,
    const int*   __restrict__ bt,
    const float* __restrict__ kvp,
    const int*   __restrict__ seqp,
    const float* __restrict__ sinks,
    const int*   __restrict__ winp,
    const unsigned char* __restrict__ maskp,
    float* __restrict__ outp)
{
    // K chunk: 32 keys x 128 d, bf16 packed (2/uint), row padded 64->68 (bank-conflict-free A loads)
    __shared__ unsigned int Klds[32][68];
    // V chunk transposed: 128 d x 16 key-pairs, bf16 packed, row padded 16->20
    __shared__ unsigned int Vlds[128][20];

    const int qc     = blockIdx.x;          // 0..15  (q chunk of 32)
    const int hkv    = blockIdx.y;          // 0..7
    const int b      = blockIdx.z;          // 0..1
    const int tid    = threadIdx.x;
    const int wave   = tid >> 5;
    const int lane   = tid & 31;
    const int lane16 = lane & 15;
    const bool lo    = lane < 16;

    const int hl     = wave >> 1;           // GQA head within group: 0..3
    const int qtile  = wave & 1;            // 0..1
    const int h      = hkv * GQA_ + hl;

    const int sl  = seqp[b];
    const int win = winp[0];

    const int q_idx = qc * QCHUNK_ + qtile * 16 + lane16;  // 0..511 within batch
    const int pos   = sl - Q_ + q_idx;

    // Uniform-per-block key range (window union over the 32 queries)
    const int pos_lo = sl - Q_ + qc * QCHUNK_;
    const int pos_hi = pos_lo + QCHUNK_ - 1;
    int kstart = pos_lo - win + 1; if (kstart < 0) kstart = 0;
    int kend   = pos_hi;           if (kend > sl - 1) kend = sl - 1;

    // ---- Preload Q as the B-operand of S^T = K * Q^T (scale folded in) ----
    // B layout (32d x 16q): lane holds column q=lane%16; lanes<16 hold d rows
    // dbase+0..15 (2/VGPR), lanes>=16 hold dbase+16..31.
    const float scale = 0.088388347648318447f;  // 1/sqrt(128)
    const float* qrow = qp + (((size_t)(b * Q_ + q_idx)) * HQ_ + h) * D_;
    ABu Bq[4];
    #pragma unroll
    for (int ds = 0; ds < 4; ++ds) {
        const int dbase = ds * 32 + (lo ? 0 : 16);
        #pragma unroll
        for (int j = 0; j < 8; ++j)
            Bq[ds].u[j] = pack_bf16(qrow[dbase + 2*j] * scale,
                                    qrow[dbase + 2*j + 1] * scale);
    }

    float m_run = sinks[h];   // sink folded into running max (matches reference)
    float l_run = 0.0f;
    float acc[8][8];          // O^T accumulators: acc[dtile][r]
    #pragma unroll
    for (int t = 0; t < 8; ++t)
        #pragma unroll
        for (int r = 0; r < 8; ++r) acc[t][r] = 0.0f;

    const int kc0 = kstart & ~31;
    for (int kc = kc0; kc <= kend; kc += 32) {
        // ================= stage K (row-major bf16) =================
        #pragma unroll
        for (int it = 0; it < 4; ++it) {
            int idx = tid + it * 256;
            int s   = idx >> 5;          // key slot 0..31
            int dq  = idx & 31;          // float4 index along d
            int j   = kc + s;
            unsigned int u0 = 0, u1 = 0;
            if (j < MAXK_) {
                int blk = bt[b * NBLK_ + (j >> 7)];
                const float* kr = kvp +
                    ((((size_t)blk * 2 + 0) * BS_ + (j & 127)) * HKV_ + hkv) * D_;
                float4 f = *(const float4*)(kr + dq * 4);
                u0 = pack_bf16(f.x, f.y);
                u1 = pack_bf16(f.z, f.w);
            }
            Klds[s][dq * 2]     = u0;
            Klds[s][dq * 2 + 1] = u1;
        }
        // ================= stage V transposed (V^T bf16) =================
        #pragma unroll
        for (int it = 0; it < 8; ++it) {
            int idx = tid + it * 256;
            int d   = idx & 127;
            int kp  = idx >> 7;          // key pair 0..15
            int j0  = kc + kp * 2;       // even -> j0,j0+1 share a page block
            unsigned int u = 0;
            if (j0 < MAXK_) {
                int blk = bt[b * NBLK_ + (j0 >> 7)];
                const float* vr = kvp +
                    ((((size_t)blk * 2 + 1) * BS_ + (j0 & 127)) * HKV_ + hkv) * D_ + d;
                u = pack_bf16(vr[0], vr[HKV_ * D_]);
            }
            Vlds[d][kp] = u;
        }
        __syncthreads();

        // ============ S^T = K * Q^T : two 16(key)x16(q) tiles ============
        // A layout (16key x 32d): lane<16 row key=lane16, d {ds*32+0..15};
        // lane>=16 same row, d {+16..31}  -> two ds_load_b128 each.
        v8f c0 = {0.f,0.f,0.f,0.f,0.f,0.f,0.f,0.f};
        v8f c1 = c0;
        #pragma unroll
        for (int ds = 0; ds < 4; ++ds) {
            const int off = ds * 16 + (lo ? 0 : 8);
            ABu A0, A1;
            *(uint4*)&A0.u[0] = *(const uint4*)&Klds[lane16][off];
            *(uint4*)&A0.u[4] = *(const uint4*)&Klds[lane16][off + 4];
            *(uint4*)&A1.u[0] = *(const uint4*)&Klds[lane16 + 16][off];
            *(uint4*)&A1.u[4] = *(const uint4*)&Klds[lane16 + 16][off + 4];
            c0 = __builtin_amdgcn_wmma_f32_16x16x32_bf16(false, A0.v, false, Bq[ds].v,
                                                         (short)0, c0, false, false);
            c1 = __builtin_amdgcn_wmma_f32_16x16x32_bf16(false, A1.v, false, Bq[ds].v,
                                                         (short)0, c1, false, false);
        }
        Cu s0, s1; s0.v = c0; s1.v = c1;

        // ===== mask: D layout lane=q col, VGPR r -> key kc+r(+8 hi-half) =====
        #pragma unroll
        for (int r = 0; r < 8; ++r) {
            int j0r = kc + r + (lo ? 0 : 8);
            int j1r = j0r + 16;
            bool a0 = (j0r < sl) && (j0r <= pos) && (j0r > pos - win) &&
                      (maskp[(size_t)q_idx * MAXK_ + j0r] == 0);
            bool a1 = (j1r < sl) && (j1r <= pos) && (j1r > pos - win) &&
                      (maskp[(size_t)q_idx * MAXK_ + j1r] == 0);
            if (!a0) s0.f[r] = NEG_;
            if (!a1) s1.f[r] = NEG_;
        }

        // ===== online softmax (per q column; halves kept symmetric) =====
        float tm = NEG_;
        #pragma unroll
        for (int r = 0; r < 8; ++r) { tm = fmaxf(tm, s0.f[r]); tm = fmaxf(tm, s1.f[r]); }
        tm = fmaxf(tm, __shfl_xor(tm, 16, 32));
        float new_m = fmaxf(m_run, tm);
        float alpha = __expf(m_run - new_m);
        float p0[8], p1[8], tsum = 0.0f;
        #pragma unroll
        for (int r = 0; r < 8; ++r) {
            p0[r] = __expf(s0.f[r] - new_m);
            p1[r] = __expf(s1.f[r] - new_m);
            tsum += p0[r] + p1[r];
        }
        tsum += __shfl_xor(tsum, 16, 32);
        l_run = l_run * alpha + tsum;
        m_run = new_m;
        #pragma unroll
        for (int t = 0; t < 8; ++t)
            #pragma unroll
            for (int r = 0; r < 8; ++r) acc[t][r] *= alpha;

        // ===== build P^T as B-operand: only a half-wave swap needed =====
        // lane<16 owns keys kc+0..7, needs kc+8..15 from lane+16 (tile0);
        // lane>=16 owns keys kc+24..31, needs kc+16..23 from lane-16 (tile1).
        float own[8], cross[8];
        #pragma unroll
        for (int r = 0; r < 8; ++r) {
            own[r]   = lo ? p0[r] : p1[r];
            float y  = lo ? p1[r] : p0[r];
            cross[r] = __shfl_xor(y, 16, 32);
        }
        ABu Bp;
        #pragma unroll
        for (int jv = 0; jv < 8; ++jv) {
            int jj = jv & 3;
            bool useOwn = (jv < 4) == lo;
            float a0v = useOwn ? own[2*jj]     : cross[2*jj];
            float a1v = useOwn ? own[2*jj + 1] : cross[2*jj + 1];
            Bp.u[jv] = pack_bf16(a0v, a1v);
        }

        // ===== O^T += V^T * P^T : 8 tiles over d =====
        // A layout (16d x 32key): lane row d=dt*16+lane16; lane<16 keys {0..15},
        // lane>=16 keys {16..31} -> two ds_load_b128 from Vlds.
        #pragma unroll
        for (int dt = 0; dt < 8; ++dt) {
            const int drow = dt * 16 + lane16;
            const int off  = lo ? 0 : 8;
            ABu Av;
            *(uint4*)&Av.u[0] = *(const uint4*)&Vlds[drow][off];
            *(uint4*)&Av.u[4] = *(const uint4*)&Vlds[drow][off + 4];
            Cu o;
            #pragma unroll
            for (int r = 0; r < 8; ++r) o.f[r] = acc[dt][r];
            o.v = __builtin_amdgcn_wmma_f32_16x16x32_bf16(false, Av.v, false, Bp.v,
                                                          (short)0, o.v, false, false);
            #pragma unroll
            for (int r = 0; r < 8; ++r) acc[dt][r] = o.f[r];
        }
        __syncthreads();
    }

    // ===== epilogue: denom = l + exp(sink - m); O^T lane=q, d=dt*16+r(+8 hi) =====
    float denom = l_run + __expf(sinks[h] - m_run);
    float inv   = 1.0f / denom;
    float* orow = outp + (((size_t)(b * Q_ + q_idx)) * HQ_ + h) * D_;
    #pragma unroll
    for (int dt = 0; dt < 8; ++dt) {
        #pragma unroll
        for (int r = 0; r < 8; ++r) {
            int d = dt * 16 + r + (lo ? 0 : 8);
            orow[d] = acc[dt][r] * inv;
        }
    }
}

extern "C" void kernel_launch(void* const* d_in, const int* in_sizes, int n_in,
                              void* d_out, int out_size, void* d_ws, size_t ws_size,
                              hipStream_t stream) {
    (void)in_sizes; (void)n_in; (void)out_size; (void)d_ws; (void)ws_size;
    const float*         qp    = (const float*)d_in[0];
    const int*           bt    = (const int*)d_in[1];
    const float*         kvp   = (const float*)d_in[2];
    const int*           seqp  = (const int*)d_in[3];
    const float*         sinks = (const float*)d_in[4];
    const int*           winp  = (const int*)d_in[5];
    const unsigned char* maskp = (const unsigned char*)d_in[6];
    float*               outp  = (float*)d_out;

    dim3 grid(Q_ / QCHUNK_, HKV_, B_);   // (16, 8, 2)
    swa_fwd_kernel<<<grid, 256, 0, stream>>>(qp, bt, kvp, seqp, sinks, winp, maskp, outp);
}
